// LocalizationNetwork_60026462929031
// MI455X (gfx1250) — compile-verified, matrix operand order checked by reading, not run on hardware
//
#include <hip/hip_runtime.h>
#include <hip/hip_bf16.h>

typedef __attribute__((ext_vector_type(16))) _Float16 v16h;
typedef __attribute__((ext_vector_type(8)))  _Float16 v8h;
typedef __attribute__((ext_vector_type(8)))  float    v8f;

#define NEG_SLOPE 0.01f

// ---------------------------------------------------------------------------
// Convert f32 OIHW conv weights to zero-padded f16 GEMM matrix [Mpad][Kpad].
// OIHW flat index = m*(Cin*9) + k  where k = cin*9 + ky*3 + kx.
// ---------------------------------------------------------------------------
__global__ void prep_weights_kernel(const float* __restrict__ w,
                                    _Float16* __restrict__ wh,
                                    int Cout, int CinK, int Mpad, int Kpad) {
    int idx = blockIdx.x * blockDim.x + threadIdx.x;
    int total = Mpad * Kpad;
    if (idx >= total) return;
    int m = idx / Kpad;
    int k = idx - m * Kpad;
    float v = (m < Cout && k < CinK) ? w[m * CinK + k] : 0.0f;
    wh[idx] = (_Float16)v;
}

// ---------------------------------------------------------------------------
// Fused conv3x3(pad1) -> avgpool2 -> leakyrelu, one layer, LDS-staged WMMA.
//
// 128 threads = 4 waves. Block covers 8 pre-pool rows x 16 pre-pool cols,
// one batch, one 16-channel cout tile. Wave w owns pre-pool rows 2w,2w+1
// (= pooled row blockIdx.y*4+w).
//
// Pipeline per block:
//   1) stage raw input tile (Cin x 10 x 20, f16, zero halo) in LDS
//   2) build pixel-major im2col tile s_b[128][Kpad] in LDS (branch-free)
//   3) stage 16 x Kpad f16 weight tile in LDS
//   4) K-loop: 6x ds_load_b128 + 2x v_wmma_f32_16x16x32_f16 per step
//   5) spill accumulators to LDS, 2x2 pool + bias + leaky, store
//
// Dynamic LDS carve (must match host):
//   [0, 8192)                 float s_acc[4][2][16][16]
//   [8192, +2*16*Kpad)        f16   s_wa[16][Kpad]
//   [.., +2*128*Kpad)         f16   s_b[128][Kpad]
//   [.., +2*Cin*200)          f16   s_in[Cin][10][20]
//   [.., +2*Kpad)             short s_off[Kpad]
// ---------------------------------------------------------------------------
__global__ __launch_bounds__(128)
void conv_pool_lrelu_kernel(const float* __restrict__ in,
                            const _Float16* __restrict__ wh,
                            const float* __restrict__ bias,
                            float* __restrict__ out,
                            int Cin, int Cout, int Hin, int Win,
                            int Kpad, int coutTiles) {
    extern __shared__ char smem[];
    float*    s_acc = (float*)smem;                                   // 8192 B
    _Float16* s_wa  = (_Float16*)(smem + 8192);                       // 16*Kpad
    _Float16* s_b   = s_wa + (size_t)16 * Kpad;                       // 128*Kpad
    _Float16* s_in  = s_b + (size_t)128 * Kpad;                       // Cin*200
    short*    s_off = (short*)(s_in + (size_t)Cin * 200);             // Kpad

    const int tid = threadIdx.x;
    const int lane = tid & 31;
    const int wave = tid >> 5;
    const int CinK = Cin * 9;
    const int Hp = Hin >> 1, Wp = Win >> 1;
    const int batch = blockIdx.z / coutTiles;
    const int ct = blockIdx.z - batch * coutTiles;
    const int coutBase = ct * 16;
    const int y0_blk = blockIdx.y * 8;        // first pre-pool row of block
    const int x0 = blockIdx.x * 16;           // first pre-pool col of block

    const float* inB = in + (size_t)batch * Cin * Hin * Win;

    // --- stage 1: raw input tile with zero halo (rows y0_blk-1..+8, cols x0-1..+16)
    {
        const int totalIn = Cin * 200;        // 10 rows x 20 cols per channel
        for (int idx = tid; idx < totalIn; idx += 128) {
            int cin = idx / 200;
            int rem = idx - cin * 200;
            int r = rem / 20;
            int c = rem - r * 20;
            int gy = y0_blk - 1 + r;
            int gx = x0 - 1 + c;
            float v = 0.0f;
            if (c < 18 && gy >= 0 && gy < Hin && gx >= 0 && gx < Win)
                v = inB[(size_t)cin * Hin * Win + (size_t)gy * Win + gx];
            s_in[idx] = (_Float16)v;
        }
    }
    // --- im2col offset LUT: k -> cin*200 + dy*20 + dx
    for (int k = tid; k < Kpad; k += 128) {
        int cin = k / 9;
        int r = k - cin * 9;
        int dy = r / 3 - 1;
        int dx = r % 3 - 1;
        s_off[k] = (short)(cin * 200 + dy * 20 + dx);
    }
    // --- stage 3: weight tile
    {
        const _Float16* wsrc = wh + (size_t)coutBase * Kpad;
        for (int idx = tid; idx < 16 * Kpad; idx += 128) s_wa[idx] = wsrc[idx];
    }
    __syncthreads();

    // --- stage 2: pixel-major im2col tile, one pixel per thread (branch-free)
    {
        // pixel tid: wave w = tid>>5, q = tid&31, tile t = q>>4, col n = q&15
        int w = tid >> 5;
        int q = tid & 31;
        int t = q >> 4;
        int n = q & 15;
        int pixBase = (w * 2 + t + 1) * 20 + (n + 1);   // local coords in s_in
        _Float16* dst = s_b + (size_t)tid * Kpad;
        for (int k = 0; k < Kpad; ++k) {
            _Float16 v = (_Float16)0.0f;
            if (k < CinK) v = s_in[pixBase + s_off[k]];
            dst[k] = v;
        }
    }
    __syncthreads();

    // --- stage 4: WMMA GEMM. 16-bit wave32 A/B layout: lane holds two
    // contiguous 8-half runs at K = baseK+[0..7] and baseK+16+[0..7].
    const int n = lane & 15;
    const int baseK = (lane < 16) ? 0 : 8;
    const _Float16* sa  = s_wa + (size_t)n * Kpad;               // A row m = lane&15
    const _Float16* sb0 = s_b + (size_t)(wave * 32 + n) * Kpad;  // tile0 pixel
    const _Float16* sb1 = sb0 + (size_t)16 * Kpad;               // tile1 pixel

    v8f acc0 = {};
    v8f acc1 = {};
    for (int ks = 0; ks < Kpad; ks += 32) {
        int o = ks + baseK;                 // multiple of 8 halves -> 16B aligned
        v8h alo  = *(const v8h*)(sa + o);
        v8h ahi  = *(const v8h*)(sa + o + 16);
        v8h b0lo = *(const v8h*)(sb0 + o);
        v8h b0hi = *(const v8h*)(sb0 + o + 16);
        v8h b1lo = *(const v8h*)(sb1 + o);
        v8h b1hi = *(const v8h*)(sb1 + o + 16);
        v16h afrag, bf0, bf1;
#pragma unroll
        for (int i = 0; i < 8; ++i) {
            afrag[i] = alo[i]; afrag[8 + i] = ahi[i];
            bf0[i]   = b0lo[i]; bf0[8 + i]  = b0hi[i];
            bf1[i]   = b1lo[i]; bf1[8 + i]  = b1hi[i];
        }
        acc0 = __builtin_amdgcn_wmma_f32_16x16x32_f16(
            false, afrag, false, bf0, (short)0, acc0, false, false);
        acc1 = __builtin_amdgcn_wmma_f32_16x16x32_f16(
            false, afrag, false, bf1, (short)0, acc1, false, false);
    }

    // --- stage 5: spill per documented C/D layout (VGPR r -> M=r+8*(lane>=16))
    {
        int chBase = (lane >= 16) ? 8 : 0;
#pragma unroll
        for (int r = 0; r < 8; ++r) {
            s_acc[(((wave * 2 + 0) * 16) + chBase + r) * 16 + n] = acc0[r];
            s_acc[(((wave * 2 + 1) * 16) + chBase + r) * 16 + n] = acc1[r];
        }
    }
    __syncthreads();

    // 2x2 mean pool + bias + leaky relu + store pooled row (pr = blockIdx.y*4+wave)
    const int pr = blockIdx.y * 4 + wave;
    const float* t0 = s_acc + (size_t)(wave * 2 + 0) * 256;
    const float* t1 = s_acc + (size_t)(wave * 2 + 1) * 256;
    for (int idx = lane; idx < 128; idx += 32) {
        int ch = idx >> 3;
        int j = idx & 7;
        int c = coutBase + ch;
        if (c < Cout) {
            float v = 0.25f * (t0[ch * 16 + 2 * j] + t0[ch * 16 + 2 * j + 1] +
                               t1[ch * 16 + 2 * j] + t1[ch * 16 + 2 * j + 1]);
            v += bias[c];
            v = (v >= 0.0f) ? v : NEG_SLOPE * v;
            out[(((size_t)batch * Cout + c) * Hp + pr) * Wp + (x0 >> 1) + j] = v;
        }
    }
}

// ---------------------------------------------------------------------------
// (32,4,8,8) -> avgpool2 -> (32,4,4,4) flattened to (32,64)
// ---------------------------------------------------------------------------
__global__ void final_pool_flatten_kernel(const float* __restrict__ in,
                                          float* __restrict__ feat) {
    int b = blockIdx.x;
    int t = threadIdx.x;            // 64 threads: c*16 + y*4 + x
    int c = t >> 4;
    int rem = t & 15;
    int y = rem >> 2;
    int x = rem & 3;
    const float* p = in + (((size_t)b * 4 + c) * 8 + y * 2) * 8 + x * 2;
    feat[b * 64 + t] = 0.25f * (p[0] + p[1] + p[8] + p[9]);
}

// ---------------------------------------------------------------------------
// gi[u,b,j] = feat[b,:] . wih[u,j,:] + bih[u,j]   (u=100, b=32, j=72, in=64)
// ---------------------------------------------------------------------------
__global__ void gru_gi_kernel(const float* __restrict__ feat,
                              const float* __restrict__ wih,
                              const float* __restrict__ bih,
                              float* __restrict__ gi) {
    int u = blockIdx.x;
    int b = blockIdx.y;
    int j = threadIdx.x;            // 72 threads
    const float* wrow = wih + ((size_t)u * 72 + j) * 64;
    const float* xr = feat + b * 64;
    float s = bih[u * 72 + j];
#pragma unroll 8
    for (int i = 0; i < 64; ++i) s += xr[i] * wrow[i];
    gi[((size_t)u * 32 + b) * 72 + j] = s;
}

// ---------------------------------------------------------------------------
// Sequential chain of 100 GRU cells + final FC -> theta (32,6).
// One block, 768 threads (tid -> (b, hh)), hidden state in LDS.
// ---------------------------------------------------------------------------
__global__ __launch_bounds__(768)
void gru_seq_fc_kernel(const float* __restrict__ gi,
                       const float* __restrict__ whh,
                       const float* __restrict__ bhh,
                       const float* __restrict__ h0,
                       const float* __restrict__ fc_w,
                       const float* __restrict__ fc_b,
                       float* __restrict__ theta) {
    __shared__ float h[32 * 24];
    int tid = threadIdx.x;
    int b = tid / 24;
    int hh = tid - b * 24;
    h[tid] = h0[hh];
    __syncthreads();

    float hrow[24];
    for (int u = 0; u < 100; ++u) {
#pragma unroll
        for (int k = 0; k < 24; ++k) hrow[k] = h[b * 24 + k];
        const float* W = whh + (size_t)u * 72 * 24;
        float dr = bhh[u * 72 + hh];
        float dz = bhh[u * 72 + 24 + hh];
        float dn = bhh[u * 72 + 48 + hh];
#pragma unroll
        for (int k = 0; k < 24; ++k) {
            dr += hrow[k] * W[hh * 24 + k];
            dz += hrow[k] * W[(24 + hh) * 24 + k];
            dn += hrow[k] * W[(48 + hh) * 24 + k];
        }
        const float* g = gi + ((size_t)u * 32 + b) * 72;
        float r = 1.0f / (1.0f + __expf(-(g[hh] + dr)));
        float z = 1.0f / (1.0f + __expf(-(g[24 + hh] + dz)));
        float n = tanhf(g[48 + hh] + r * dn);
        float hn = (1.0f - z) * n + z * hrow[hh];
        __syncthreads();
        h[tid] = hn;
        __syncthreads();
    }

    if (hh < 6) {
        float s = fc_b[hh];
#pragma unroll
        for (int k = 0; k < 24; ++k) s += h[b * 24 + k] * fc_w[hh * 24 + k];
        theta[b * 6 + hh] = s;
    }
}

// ---------------------------------------------------------------------------
// Affine grid + bilinear grid_sample (align_corners=False, zeros padding).
// img: (32,1,512,512), theta: (32,6), out: (32,1,512,512)
// ---------------------------------------------------------------------------
__global__ void grid_sample_kernel(const float* __restrict__ img,
                                   const float* __restrict__ theta,
                                   float* __restrict__ out) {
    const int W = 512, H = 512;
    long long idx = (long long)blockIdx.x * blockDim.x + threadIdx.x;
    long long total = 32LL * H * W;
    if (idx >= total) return;
    int x = (int)(idx % W);
    long long t = idx / W;
    int y = (int)(t % H);
    int b = (int)(t / H);

    const float* th = theta + b * 6;
    const float* im = img + (long long)b * H * W;
    __builtin_prefetch(im, 0, 1);   // global_prefetch_b8

    float xs = (2.0f * x + 1.0f) / W - 1.0f;
    float ys = (2.0f * y + 1.0f) / H - 1.0f;
    float gx = th[0] * xs + th[1] * ys + th[2];
    float gy = th[3] * xs + th[4] * ys + th[5];

    float ix = ((gx + 1.0f) * W - 1.0f) * 0.5f;
    float iy = ((gy + 1.0f) * H - 1.0f) * 0.5f;
    float x0f = floorf(ix), y0f = floorf(iy);
    float wx1 = ix - x0f, wy1 = iy - y0f;
    float wx0 = 1.0f - wx1, wy0 = 1.0f - wy1;
    int x0 = (int)x0f, y0 = (int)y0f;
    int x1 = x0 + 1, y1 = y0 + 1;

    float v00 = 0.0f, v01 = 0.0f, v10 = 0.0f, v11 = 0.0f;
    if (y0 >= 0 && y0 < H) {
        if (x0 >= 0 && x0 < W) v00 = im[(long long)y0 * W + x0];
        if (x1 >= 0 && x1 < W) v01 = im[(long long)y0 * W + x1];
    }
    if (y1 >= 0 && y1 < H) {
        if (x0 >= 0 && x0 < W) v10 = im[(long long)y1 * W + x0];
        if (x1 >= 0 && x1 < W) v11 = im[(long long)y1 * W + x1];
    }
    out[idx] = v00 * wy0 * wx0 + v01 * wy0 * wx1 + v10 * wy1 * wx0 + v11 * wy1 * wx1;
}

// ---------------------------------------------------------------------------
// Host-side orchestration
// ---------------------------------------------------------------------------
extern "C" void kernel_launch(void* const* d_in, const int* in_sizes, int n_in,
                              void* d_out, int out_size, void* d_ws, size_t ws_size,
                              hipStream_t stream) {
    (void)in_sizes; (void)n_in; (void)out_size; (void)ws_size;

    const float* x = (const float*)d_in[0];
    const float* conv_w[6];
    const float* conv_b[6];
    for (int i = 0; i < 6; ++i) {
        conv_w[i] = (const float*)d_in[1 + 2 * i];
        conv_b[i] = (const float*)d_in[2 + 2 * i];
    }
    const float* gru_wih = (const float*)d_in[13];
    const float* gru_whh = (const float*)d_in[14];
    const float* gru_bih = (const float*)d_in[15];
    const float* gru_bhh = (const float*)d_in[16];
    const float* h0      = (const float*)d_in[17];
    const float* fc_w    = (const float*)d_in[18];
    const float* fc_b    = (const float*)d_in[19];

    // Workspace layout
    char* ws = (char*)d_ws;
    const size_t OFF_BUF0 = 0;                          // 32*10*256*256*4 = 83,886,080 B
    const size_t OFF_BUF1 = 83886080;                   // 32*20*128*128*4 = 41,943,040 B
    const size_t OFF_W16  = OFF_BUF1 + 41943040;        // padded f16 weights, 47,104 B
    const size_t OFF_FEAT = OFF_W16 + 47104;            // 32*64*4 = 8,192 B
    const size_t OFF_GI   = OFF_FEAT + 8192;            // 100*32*72*4 = 921,600 B
    const size_t OFF_TH   = OFF_GI + 921600;            // 32*6*4 B

    float* buf0  = (float*)(ws + OFF_BUF0);
    float* buf1  = (float*)(ws + OFF_BUF1);
    _Float16* w16base = (_Float16*)(ws + OFF_W16);
    float* feat  = (float*)(ws + OFF_FEAT);
    float* gi    = (float*)(ws + OFF_GI);
    float* theta = (float*)(ws + OFF_TH);

    // Per-layer geometry
    const int Cin[6]  = {1, 10, 20, 30, 20, 10};
    const int Cout[6] = {10, 20, 30, 20, 10, 4};
    const int Hin[6]  = {512, 256, 128, 64, 32, 16};
    const float* lin[6]  = {x, buf0, buf1, buf0, buf1, buf0};
    float*       lout[6] = {buf0, buf1, buf0, buf1, buf0, buf1};

    _Float16* w16[6];
    {
        size_t off = 0;
        for (int l = 0; l < 6; ++l) {
            int cT = (Cout[l] + 15) / 16;
            int Mpad = cT * 16;
            int Kpad = ((Cin[l] * 9 + 31) / 32) * 32;
            w16[l] = w16base + off;
            off += (size_t)Mpad * Kpad;
        }
    }

    // Allow large dynamic LDS for the conv kernel (WGP has 320 KB).
    hipFuncSetAttribute((const void*)conv_pool_lrelu_kernel,
                        hipFuncAttributeMaxDynamicSharedMemorySize, 114688);

    // 1) weight prep (f32 -> padded f16 GEMM matrices)
    for (int l = 0; l < 6; ++l) {
        int cT = (Cout[l] + 15) / 16;
        int Mpad = cT * 16;
        int Kpad = ((Cin[l] * 9 + 31) / 32) * 32;
        int total = Mpad * Kpad;
        prep_weights_kernel<<<(total + 255) / 256, 256, 0, stream>>>(
            conv_w[l], w16[l], Cout[l], Cin[l] * 9, Mpad, Kpad);
    }

    // 2) six fused conv+pool+leaky layers (LDS-staged WMMA)
    for (int l = 0; l < 6; ++l) {
        int cT = (Cout[l] + 15) / 16;
        int Kpad = ((Cin[l] * 9 + 31) / 32) * 32;
        // dynamic LDS: acc 8192 + f16 weights 16*Kpad + im2col 128*Kpad
        //              + input tile Cin*200 (f16) + offset LUT Kpad (i16)
        size_t shmem = 8192 + (size_t)2 * Kpad * (16 + 128)
                     + (size_t)Cin[l] * 400 + (size_t)2 * Kpad;
        dim3 grid(Hin[l] / 16, Hin[l] / 8, 32 * cT);
        conv_pool_lrelu_kernel<<<grid, 128, shmem, stream>>>(
            lin[l], w16[l], conv_b[l], lout[l],
            Cin[l], Cout[l], Hin[l], Hin[l], Kpad, cT);
    }

    // 3) adaptive pool 8x8 -> 4x4, flatten to (32,64)
    final_pool_flatten_kernel<<<32, 64, 0, stream>>>(lout[5], feat);

    // 4) input-side GRU gates for all 100 units
    {
        dim3 grid(100, 32);
        gru_gi_kernel<<<grid, 72, 0, stream>>>(feat, gru_wih, gru_bih, gi);
    }

    // 5) sequential 100-cell GRU chain + FC -> theta
    gru_seq_fc_kernel<<<1, 768, 0, stream>>>(gi, gru_whh, gru_bhh, h0, fc_w, fc_b, theta);

    // 6) affine grid + bilinear sample -> output
    {
        long long total = 32LL * 512 * 512;
        grid_sample_kernel<<<(unsigned)((total + 255) / 256), 256, 0, stream>>>(
            x, theta, (float*)d_out);
    }
}
